// RNA_Encoder_42545946034710
// MI455X (gfx1250) — compile-verified, matrix operand order checked by reading, not run on hardware
//
#include <hip/hip_runtime.h>
#include <stdint.h>

#define N_NODES 10000
#define IN_DIM  3000
#define HIDDEN  512
#define HEADS   4
#define NHTOT   (N_NODES * HEADS)
#define N_EDGES 60000
#define TOT_E   (N_EDGES + N_NODES)   // with self loops
#define NOUT    (HEADS * HIDDEN)      // 2048
#define K_PAD   3008                  // 3000 padded to mult of 32
#define BM      128
#define BN      128
#define KSTEP   32
#define NK      (K_PAD / KSTEP)       // 94
#define M_TILES 79
#define M_PAD   (M_TILES * BM)        // 10112
#define N_TILES (NOUT / BN)           // 16
#define NEG_SLOPE 0.2f
#define LN_EPS    1e-5f
#define LDSA_STR  5                   // uint4 row stride (4 data + 1 pad) -> conflict-free ds_load_b128

typedef __attribute__((ext_vector_type(16))) __bf16 v16bf;
typedef __attribute__((ext_vector_type(8)))  float  v8f;
typedef int v4i __attribute__((vector_size(16)));
typedef __attribute__((address_space(1))) v4i* gptr4;
typedef __attribute__((address_space(3))) v4i* lptr4;

#if __has_builtin(__builtin_amdgcn_global_load_async_to_lds_b128)
#define USE_ASYNC 1
#endif

// copy 16 bytes global->LDS (async on CDNA5 if available)
__device__ __forceinline__ void cp16(const uint4* g, uint4* l) {
#ifdef USE_ASYNC
  __builtin_amdgcn_global_load_async_to_lds_b128(
      (gptr4)const_cast<uint4*>(g), (lptr4)l, 0, 0);
#else
  *l = *g;
#endif
}
__device__ __forceinline__ void wait_async() {
#ifdef USE_ASYNC
#if __has_builtin(__builtin_amdgcn_s_wait_asynccnt)
  __builtin_amdgcn_s_wait_asynccnt(0);
#else
  asm volatile("s_wait_asynccnt 0x0" ::: "memory");
#endif
#endif
}

__device__ __forceinline__ unsigned short f2bf(float f) {
  unsigned u = __float_as_uint(f);
  unsigned r = u + 0x7fffu + ((u >> 16) & 1u);   // round-to-nearest-even
  return (unsigned short)(r >> 16);
}
__device__ __forceinline__ unsigned fkey(float f) {  // monotone float->u32
  unsigned b = __float_as_uint(f);
  return (b & 0x80000000u) ? ~b : (b | 0x80000000u);
}
__device__ __forceinline__ float funkey(unsigned u) {
  unsigned b = (u & 0x80000000u) ? (u & 0x7fffffffu) : ~u;
  return __uint_as_float(b);
}
__device__ __forceinline__ void edge_ends(const int* ei, int e, int& s, int& d) {
  if (e < N_EDGES) { s = ei[e]; d = ei[N_EDGES + e]; }
  else             { s = e - N_EDGES; d = s; }
}

// ---- bf16 conversion / padding ----
__global__ void k_conv_x(const float* __restrict__ x, unsigned short* __restrict__ xb) {
  long long i = (long long)blockIdx.x * blockDim.x + threadIdx.x;
  if (i >= (long long)M_PAD * K_PAD) return;
  int row = (int)(i / K_PAD), col = (int)(i % K_PAD);
  float v = (row < N_NODES && col < IN_DIM) ? x[(size_t)row * IN_DIM + col] : 0.f;
  xb[i] = f2bf(v);
}
__global__ void k_conv_w(const float* __restrict__ W, unsigned short* __restrict__ wt) {
  long long i = (long long)blockIdx.x * blockDim.x + threadIdx.x;
  if (i >= (long long)NOUT * K_PAD) return;
  int n = (int)(i / K_PAD), k = (int)(i % K_PAD);
  float v = (k < IN_DIM) ? W[(size_t)k * NOUT + n] : 0.f;   // transpose: Wt[n][k]
  wt[i] = f2bf(v);
}

// ---- WMMA GEMM: h[M,2048] = xb[M,K] * Wt[N,K]^T ----
// 128x128 block tile, 8 wave32 as 4(M)x2(N), 32x64 per wave, double-buffered
// LDS with async global->LDS copies.
__global__ __launch_bounds__(256) void k_gemm(const unsigned short* __restrict__ xb,
                                              const unsigned short* __restrict__ wt,
                                              float* __restrict__ hout) {
  __shared__ uint4 sA[2][BM * LDSA_STR];   // 128 rows x 32 bf16 (+pad), x2 buffers
  __shared__ uint4 sB[2][BN * LDSA_STR];
  const int tid   = threadIdx.x;
  const int lane  = tid & 31;
  const int wid   = tid >> 5;
  const int waveM = wid & 3;            // 4 waves along M
  const int waveN = wid >> 2;           // 2 waves along N
  const int m0 = blockIdx.x * BM;
  const int n0 = blockIdx.y * BN;
  const uint4* gA = (const uint4*)xb;
  const uint4* gB = (const uint4*)wt;

  // per-thread staging chunks: 512 A-chunks + 512 B-chunks, 4 per thread
  const int ca0_row = tid >> 2,         ca0_cc = tid & 3;
  const int ca1_row = (tid + 256) >> 2, ca1_cc = (tid + 256) & 3;

  auto issueTile = [&](int k, int buf) {
    cp16(gA + ((size_t)(m0 + ca0_row) * K_PAD + k) / 8 + ca0_cc,
         &sA[buf][ca0_row * LDSA_STR + ca0_cc]);
    cp16(gA + ((size_t)(m0 + ca1_row) * K_PAD + k) / 8 + ca1_cc,
         &sA[buf][ca1_row * LDSA_STR + ca1_cc]);
    cp16(gB + ((size_t)(n0 + ca0_row) * K_PAD + k) / 8 + ca0_cc,
         &sB[buf][ca0_row * LDSA_STR + ca0_cc]);
    cp16(gB + ((size_t)(n0 + ca1_row) * K_PAD + k) / 8 + ca1_cc,
         &sB[buf][ca1_row * LDSA_STR + ca1_cc]);
  };

  v8f acc[2][4] = {};
  const int hsel = lane >> 4;           // which 8-wide K chunk this lane holds
  const int r    = lane & 15;           // matrix row within fragment

  issueTile(0, 0);
  for (int kk = 0; kk < NK; ++kk) {
    const int buf = kk & 1;
    wait_async();                        // my tile-kk copies done
    __syncthreads();                     // everyone's tile-kk copies done;
                                         // everyone done reading buf^1
    if (kk + 1 < NK) issueTile((kk + 1) * KSTEP, buf ^ 1);

    union FB { uint4 u[2]; v16bf v; } a[2], b[4];
    #pragma unroll
    for (int mi = 0; mi < 2; ++mi) {
      int base = (waveM * 32 + mi * 16 + r) * LDSA_STR;
      a[mi].u[0] = sA[buf][base + hsel];
      a[mi].u[1] = sA[buf][base + 2 + hsel];
    }
    #pragma unroll
    for (int ni = 0; ni < 4; ++ni) {
      int base = (waveN * 64 + ni * 16 + r) * LDSA_STR;
      b[ni].u[0] = sB[buf][base + hsel];
      b[ni].u[1] = sB[buf][base + 2 + hsel];
    }
    #pragma unroll
    for (int mi = 0; mi < 2; ++mi)
      #pragma unroll
      for (int ni = 0; ni < 4; ++ni)
        acc[mi][ni] = __builtin_amdgcn_wmma_f32_16x16x32_bf16(
            false, a[mi].v, false, b[ni].v, (short)0, acc[mi][ni], false, false);
  }

  // C/D layout: vgpr rr, lanes 0-15 -> M=rr, lanes 16-31 -> M=rr+8; N = lane&15
  const int hs = lane >> 4;
  for (int mi = 0; mi < 2; ++mi)
    for (int ni = 0; ni < 4; ++ni) {
      int col = n0 + waveN * 64 + ni * 16 + r;
      #pragma unroll
      for (int rr = 0; rr < 8; ++rr) {
        int row = m0 + waveM * 32 + mi * 16 + rr + 8 * hs;
        if (row < N_NODES) hout[(size_t)row * NOUT + col] = acc[mi][ni][rr];
      }
    }
}

// ---- per-node attention logits: wave per (node, head) ----
__global__ __launch_bounds__(256) void k_logits(const float* __restrict__ h,
    const float* __restrict__ att_src, const float* __restrict__ att_dst,
    float* __restrict__ asrc, float* __restrict__ adst) {
  int g = (int)((blockIdx.x * 256 + threadIdx.x) >> 5);
  int lane = threadIdx.x & 31;
  if (g >= NHTOT) return;
  int n = g >> 2, hd = g & 3;
  const float* hp = h + (size_t)n * NOUT + hd * HIDDEN;
  const float* as = att_src + hd * HIDDEN;
  const float* ad = att_dst + hd * HIDDEN;
  float s1 = 0.f, s2 = 0.f;
  for (int c = lane; c < HIDDEN; c += 32) {
    float hv = hp[c];
    s1 += hv * as[c];
    s2 += hv * ad[c];
  }
  for (int m = 16; m; m >>= 1) { s1 += __shfl_xor(s1, m, 32); s2 += __shfl_xor(s2, m, 32); }
  if (lane == 0) { asrc[g] = s1; adst[g] = s2; }
}

__global__ void k_init_nodes(unsigned* __restrict__ nmax, float* __restrict__ nsum) {
  int i = blockIdx.x * blockDim.x + threadIdx.x;
  if (i < NHTOT) { nmax[i] = 0x007fffffu; /* fkey(-inf) */ nsum[i] = 0.f; }
}
__global__ void k_zero_f(float* __restrict__ p, long long n) {
  long long i = (long long)blockIdx.x * blockDim.x + threadIdx.x;
  if (i < n) p[i] = 0.f;
}

__global__ void k_edge_max(const int* __restrict__ ei, const float* __restrict__ asrc,
                           const float* __restrict__ adst, unsigned* __restrict__ nmax) {
  int e = blockIdx.x * blockDim.x + threadIdx.x;
  if (e >= TOT_E) return;
  int s, d; edge_ends(ei, e, s, d);
  for (int hd = 0; hd < HEADS; ++hd) {
    float v = asrc[s * HEADS + hd] + adst[d * HEADS + hd];
    v = v > 0.f ? v : NEG_SLOPE * v;
    atomicMax(&nmax[d * HEADS + hd], fkey(v));
  }
}
__global__ void k_edge_sum(const int* __restrict__ ei, const float* __restrict__ asrc,
                           const float* __restrict__ adst, const unsigned* __restrict__ nmax,
                           float* __restrict__ nsum) {
  int e = blockIdx.x * blockDim.x + threadIdx.x;
  if (e >= TOT_E) return;
  int s, d; edge_ends(ei, e, s, d);
  for (int hd = 0; hd < HEADS; ++hd) {
    float v = asrc[s * HEADS + hd] + adst[d * HEADS + hd];
    v = v > 0.f ? v : NEG_SLOPE * v;
    atomicAdd(&nsum[d * HEADS + hd], expf(v - funkey(nmax[d * HEADS + hd])));
  }
}

// ---- weighted aggregation: wave per (edge, head) ----
__global__ __launch_bounds__(256) void k_edge_agg(const int* __restrict__ ei,
    const float* __restrict__ asrc, const float* __restrict__ adst,
    const unsigned* __restrict__ nmax, const float* __restrict__ nsum,
    const float* __restrict__ h, float* __restrict__ agg) {
  int g = (int)((blockIdx.x * 256 + threadIdx.x) >> 5);
  int lane = threadIdx.x & 31;
  if (g >= TOT_E * HEADS) return;
  int e = g >> 2, hd = g & 3;
  int s, d; edge_ends(ei, e, s, d);
  float v = asrc[s * HEADS + hd] + adst[d * HEADS + hd];
  v = v > 0.f ? v : NEG_SLOPE * v;
  float alpha = expf(v - funkey(nmax[d * HEADS + hd])) / (nsum[d * HEADS + hd] + 1e-16f);
  const float* hp = h + (size_t)s * NOUT + hd * HIDDEN;
  float* ap = agg + ((size_t)d * HEADS + hd) * HIDDEN;
  for (int c = lane; c < HIDDEN; c += 32)
    atomicAdd(&ap[c], alpha * hp[c]);
}

// ---- head mean + bias + ELU + LayerNorm: block per node ----
__global__ __launch_bounds__(256) void k_final(const float* __restrict__ agg,
    const float* __restrict__ bias, const float* __restrict__ gamma,
    const float* __restrict__ beta, float* __restrict__ out) {
  __shared__ float rs[256], rq[256];
  int n = blockIdx.x, tid = threadIdx.x;
  const float* ap = agg + (size_t)n * HEADS * HIDDEN;
  float v[2]; float s = 0.f, q = 0.f;
  #pragma unroll
  for (int i = 0; i < 2; ++i) {
    int c = tid + i * 256;
    float t = 0.25f * (ap[c] + ap[HIDDEN + c] + ap[2 * HIDDEN + c] + ap[3 * HIDDEN + c]) + bias[c];
    t = t > 0.f ? t : expm1f(t);
    v[i] = t; s += t; q += t * t;
  }
  rs[tid] = s; rq[tid] = q;
  __syncthreads();
  for (int off = 128; off; off >>= 1) {
    if (tid < off) { rs[tid] += rs[tid + off]; rq[tid] += rq[tid + off]; }
    __syncthreads();
  }
  float mu  = rs[0] * (1.f / HIDDEN);
  float var = rq[0] * (1.f / HIDDEN) - mu * mu;
  float inv = rsqrtf(var + LN_EPS);
  #pragma unroll
  for (int i = 0; i < 2; ++i) {
    int c = tid + i * 256;
    out[(size_t)n * HIDDEN + c] = (v[i] - mu) * inv * gamma[c] + beta[c];
  }
}

extern "C" void kernel_launch(void* const* d_in, const int* in_sizes, int n_in,
                              void* d_out, int out_size, void* d_ws, size_t ws_size,
                              hipStream_t stream) {
  (void)in_sizes; (void)n_in; (void)out_size; (void)ws_size;
  const float* x       = (const float*)d_in[0];
  const int*   ei      = (const int*)d_in[1];
  const float* W       = (const float*)d_in[2];
  const float* att_src = (const float*)d_in[3];
  const float* att_dst = (const float*)d_in[4];
  const float* bias    = (const float*)d_in[5];
  const float* gamma   = (const float*)d_in[6];
  const float* beta    = (const float*)d_in[7];
  float* out = (float*)d_out;

  char* p = (char*)d_ws;
  auto carve = [&](size_t bytes) -> char* {
    char* r = p; p += (bytes + 255) & ~(size_t)255; return r;
  };
  unsigned short* xb = (unsigned short*)carve((size_t)M_PAD * K_PAD * 2);
  unsigned short* wt = (unsigned short*)carve((size_t)NOUT * K_PAD * 2);
  float* h     = (float*)carve((size_t)N_NODES * NOUT * 4);
  float* asrc  = (float*)carve((size_t)NHTOT * 4);
  float* adst  = (float*)carve((size_t)NHTOT * 4);
  unsigned* nmax = (unsigned*)carve((size_t)NHTOT * 4);
  float* nsum  = (float*)carve((size_t)NHTOT * 4);
  float* agg   = (float*)carve((size_t)N_NODES * HEADS * HIDDEN * 4);

  { long long tot = (long long)M_PAD * K_PAD;
    k_conv_x<<<(unsigned)((tot + 255) / 256), 256, 0, stream>>>(x, xb); }
  { long long tot = (long long)NOUT * K_PAD;
    k_conv_w<<<(unsigned)((tot + 255) / 256), 256, 0, stream>>>(W, wt); }
  k_gemm<<<dim3(M_TILES, N_TILES), 256, 0, stream>>>(xb, wt, h);
  k_logits<<<(NHTOT * 32 + 255) / 256, 256, 0, stream>>>(h, att_src, att_dst, asrc, adst);
  k_init_nodes<<<(NHTOT + 255) / 256, 256, 0, stream>>>(nmax, nsum);
  { long long tot = (long long)N_NODES * HEADS * HIDDEN;
    k_zero_f<<<(unsigned)((tot + 255) / 256), 256, 0, stream>>>(agg, tot); }
  k_edge_max<<<(TOT_E + 255) / 256, 256, 0, stream>>>(ei, asrc, adst, nmax);
  k_edge_sum<<<(TOT_E + 255) / 256, 256, 0, stream>>>(ei, asrc, adst, nmax, nsum);
  { long long tot = (long long)TOT_E * HEADS * 32;
    k_edge_agg<<<(unsigned)((tot + 255) / 256), 256, 0, stream>>>(ei, asrc, adst, nmax, nsum, h, agg); }
  k_final<<<N_NODES, 256, 0, stream>>>(agg, bias, gamma, beta, out);
}
